// RelativeAttention_79302276153780
// MI455X (gfx1250) — matrix operand order repaired and checked by reading, not compiled, not run on hardware
//
#include <hip/hip_runtime.h>

#define QLEN  1024
#define KLEN  1024
#define BSZ   2
#define NHEAD 16
#define DHEAD 64
#define SCALE 0.125f   /* 1/sqrt(64) */
#define NKT   (KLEN / 64)

typedef __attribute__((ext_vector_type(16))) __bf16 v16bf;
typedef __attribute__((ext_vector_type(8)))  float  v8f;
typedef unsigned int   __attribute__((may_alias)) u32a;
typedef unsigned short __attribute__((may_alias)) u16a;

union Frag16 { unsigned int u[8]; v16bf v; };

__device__ __forceinline__ unsigned int packbf2(float a, float b) {
  unsigned int ua = __builtin_bit_cast(unsigned int, a);
  unsigned int ub = __builtin_bit_cast(unsigned int, b);
  ua += 0x7FFFu + ((ua >> 16) & 1u);          // round-to-nearest-even bf16
  ub += 0x7FFFu + ((ub >> 16) & 1u);
  return (ua >> 16) | (ub & 0xFFFF0000u);
}
__device__ __forceinline__ unsigned short f2bf(float a) {
  unsigned int ua = __builtin_bit_cast(unsigned int, a);
  ua += 0x7FFFu + ((ua >> 16) & 1u);
  return (unsigned short)(ua >> 16);
}

#define KROW 33   // words/row of ksh,krsh (32 bf16-pairs + pad)
#define VROW 34   // words/row of vsh (transposed [d][key], pad keeps pairs aligned)
#define PROW 33   // words/row of per-wave P buffer
#define MROW 68   // floats/row of mask tile (272B: 16B-aligned, bank-staggered)

__global__ __launch_bounds__(128)
void relattn_flash_kernel(const float* __restrict__ q,
                          const float* __restrict__ kh,
                          const float* __restrict__ vh,
                          const float* __restrict__ kr,
                          const float* __restrict__ seg_embed,
                          const float* __restrict__ seg_mat,
                          const float* __restrict__ rwb,
                          const float* __restrict__ rrb,
                          const float* __restrict__ rsb,
                          const float* __restrict__ amask,
                          float* __restrict__ out)
{
  __shared__ unsigned int ksh [64 * KROW];                 // K-tile  bf16 [key][d]
  __shared__ unsigned int krsh[64 * KROW];                 // Kr-tile bf16 [key][d] (j+1)
  __shared__ unsigned int vsh [64 * VROW];                 // V-tile  bf16 [d][key]
  __shared__ unsigned int pbuf[4 * 16 * PROW];             // per-wave P tile bf16
  __shared__ float        efp [64][2];                     // segment projections
  __shared__ __align__(16) float msh[2 * 64 * MROW];       // mask tiles (async, dbl-buf)

  const int qt   = blockIdx.x;
  const int bn   = blockIdx.y;
  const int b    = bn / NHEAD;
  const int n    = bn % NHEAD;
  const int tid  = threadIdx.x;
  const int w    = tid >> 5;
  const int lane = tid & 31;
  const int lh   = lane >> 4;
  const int l15  = lane & 15;

  const long long rowStride = (long long)BSZ * NHEAD * DHEAD;   // 2048
  const long long bnOff     = (long long)(b * NHEAD + n) * DHEAD;

  const unsigned mshBase = (unsigned)(size_t)&msh[0];      // LDS byte address (addr[31:0])

  // ---- kick off async mask tile 0 (double-buffered through the loop) ----
  {
    for (int k = 0; k < 2; ++k) {
      const int cid = tid * 2 + k;                 // 256 chunks of 16B
      const int row = cid >> 2, ch = cid & 3;
      const unsigned voff = (unsigned)(((qt * 64 + row) * KLEN) + ch * 4) * 4u;
      const unsigned lds  = mshBase + (unsigned)(row * MROW + ch * 4) * 4u;
      asm volatile("global_load_async_to_lds_b128 %0, %1, %2"
                   :: "v"(lds), "v"(voff), "s"(amask) : "memory");
    }
  }

  // ---- segment projection: efp[i_local][s] = (q[i]+r_s_bias) . seg_embed[s][n] ----
  {
    const int il = tid >> 1, s = tid & 1;
    const int i  = qt * 64 + il;
    const float* qp = q + (long long)i * rowStride + bnOff;
    const float* se = seg_embed + (s * NHEAD + n) * DHEAD;
    const float* rb = rsb + n * DHEAD;
    float acc = 0.f;
    for (int d = 0; d < DHEAD; ++d) acc += (qp[d] + rb[d]) * se[d];
    efp[il][s] = acc;
  }

  // ---- Q A-fragments (bf16 16x32 A-layout), two bias variants, two K-steps ----
  const int qrow_local = w * 16 + l15;
  const int i0 = qt * 64 + qrow_local;
  Frag16 aw[2], ar[2];
  {
    const float* qp = q + (long long)i0 * rowStride + bnOff;
    const float* bw = rwb + n * DHEAD;
    const float* br = rrb + n * DHEAD;
    for (int ks = 0; ks < 2; ++ks)
      for (int v = 0; v < 8; ++v) {
        const int d = ks * 32 + lh * 16 + 2 * v;
        const float a0 = qp[d], a1 = qp[d + 1];
        aw[ks].u[v] = packbf2(a0 + bw[d], a1 + bw[d + 1]);
        ar[ks].u[v] = packbf2(a0 + br[d], a1 + br[d + 1]);
      }
  }

  __syncthreads();                                 // efp ready

  float ef0[8], ef1[8], m[8], l[8];
  v8f o[4];
  for (int r = 0; r < 8; ++r) {
    const int rl = w * 16 + r + 8 * lh;
    ef0[r] = efp[rl][0];
    ef1[r] = efp[rl][1];
    m[r] = -1e30f;
    l[r] = 0.f;
  }
  for (int c = 0; c < 4; ++c)
    for (int r = 0; r < 8; ++r) o[c][r] = 0.f;

  // =================== flash loop over key tiles ===================
  for (int kt = 0; kt < NKT; ++kt) {
    __syncthreads();                               // prev tile fully consumed

    // issue async mask tile kt+1 into the other buffer, retire tile kt
    if (kt + 1 < NKT) {
      const int nbuf = (kt + 1) & 1;
      for (int k = 0; k < 2; ++k) {
        const int cid = tid * 2 + k;
        const int row = cid >> 2, ch = cid & 3;
        const unsigned voff =
            (unsigned)(((qt * 64 + row) * KLEN) + (kt + 1) * 64 + ch * 4) * 4u;
        const unsigned lds =
            mshBase + (unsigned)(nbuf * (64 * MROW) + row * MROW + ch * 4) * 4u;
        asm volatile("global_load_async_to_lds_b128 %0, %1, %2"
                     :: "v"(lds), "v"(voff), "s"(amask) : "memory");
      }
      asm volatile("s_wait_asynccnt 0x2" ::: "memory");   // tile kt (2 oldest) done
    } else {
      asm volatile("s_wait_asynccnt 0x0" ::: "memory");
    }

    // prefetch next K/V tile into caches
    if (kt + 1 < NKT) {
      const int jn = (kt + 1) * 64 + (tid >> 1);
      const float* pf = ((tid & 1) ? vh : kh) + (long long)jn * rowStride + bnOff;
      __builtin_prefetch(pf, 0, 1);
    }

    // cooperative tile load: fp32 -> bf16 into LDS
    for (int idx = tid; idx < 64 * 32; idx += 128) {
      const int row = idx >> 5;                    // key within tile
      const int pr  = idx & 31;                    // bf16-pair index along d
      const int jk  = kt * 64 + row;
      const float* kp  = kh + (long long)jk * rowStride + bnOff + pr * 2;
      ksh[row * KROW + pr] = packbf2(kp[0], kp[1]);
      const float* krp = kr + (long long)(jk + 1) * rowStride + bnOff + pr * 2; // rel-shift
      krsh[row * KROW + pr] = packbf2(krp[0], krp[1]);
      const float* vp  = vh + (long long)jk * rowStride + bnOff + pr * 2;
      u16a* vs = (u16a*)vsh;                       // transposed scatter vsh[d][key]
      vs[(pr * 2 + 0) * (VROW * 2) + row] = f2bf(vp[0]);
      vs[(pr * 2 + 1) * (VROW * 2) + row] = f2bf(vp[1]);
    }
    __syncthreads();                               // tiles + async mask visible

    // ---- scores: preload all 4 B-frags, then 4 back-to-back WMMAs ----
    v8f sc[4];
    for (int c = 0; c < 4; ++c) {
      const int jrow = c * 16 + l15;
      Frag16 bk0, bk1, br0, br1;
      for (int v = 0; v < 8; ++v) {
        bk0.u[v] = ((u32a*)ksh )[jrow * KROW +      lh * 8 + v];
        bk1.u[v] = ((u32a*)ksh )[jrow * KROW + 16 + lh * 8 + v];
        br0.u[v] = ((u32a*)krsh)[jrow * KROW +      lh * 8 + v];
        br1.u[v] = ((u32a*)krsh)[jrow * KROW + 16 + lh * 8 + v];
      }
      v8f acc;
      for (int r = 0; r < 8; ++r) acc[r] = 0.f;
      acc = __builtin_amdgcn_wmma_f32_16x16x32_bf16(false, aw[0].v, false, bk0.v,
                                                    (short)0, acc, false, false);
      acc = __builtin_amdgcn_wmma_f32_16x16x32_bf16(false, aw[1].v, false, bk1.v,
                                                    (short)0, acc, false, false);
      acc = __builtin_amdgcn_wmma_f32_16x16x32_bf16(false, ar[0].v, false, br0.v,
                                                    (short)0, acc, false, false);
      acc = __builtin_amdgcn_wmma_f32_16x16x32_bf16(false, ar[1].v, false, br1.v,
                                                    (short)0, acc, false, false);
      sc[c] = acc;
    }

    // ---- fuse segment term + mask (from async LDS tile) + scale ----
    const int mbuf = (kt & 1) * (64 * MROW);
    for (int c = 0; c < 4; ++c) {
      const int j = kt * 64 + c * 16 + l15;
      for (int r = 0; r < 8; ++r) {
        const int il = w * 16 + r + 8 * lh;
        const int i  = qt * 64 + il;
        const long long ij = (long long)i * KLEN + j;
        const float* sm = seg_mat + (ij * BSZ + b) * 2;
        const float ef   = sm[0] * ef0[r] + sm[1] * ef1[r];
        const float mval = msh[mbuf + il * MROW + c * 16 + l15];
        sc[c][r] = (sc[c][r] + ef) * SCALE - 1e30f * mval;
      }
    }

    // ---- online softmax (per row, 16-lane half reductions) ----
    for (int r = 0; r < 8; ++r) {
      float mx = fmaxf(fmaxf(sc[0][r], sc[1][r]), fmaxf(sc[2][r], sc[3][r]));
      mx = fmaxf(mx, __shfl_xor(mx, 1));
      mx = fmaxf(mx, __shfl_xor(mx, 2));
      mx = fmaxf(mx, __shfl_xor(mx, 4));
      mx = fmaxf(mx, __shfl_xor(mx, 8));
      const float mn    = fmaxf(m[r], mx);
      const float alpha = __expf(m[r] - mn);
      m[r] = mn;
      float rs = 0.f;
      for (int c = 0; c < 4; ++c) {
        const float p = __expf(sc[c][r] - mn);
        sc[c][r] = p;
        rs += p;
      }
      rs += __shfl_xor(rs, 1);
      rs += __shfl_xor(rs, 2);
      rs += __shfl_xor(rs, 4);
      rs += __shfl_xor(rs, 8);
      l[r] = l[r] * alpha + rs;
      for (int c = 0; c < 4; ++c) o[c][r] *= alpha;
    }

    // ---- P (C-layout) -> wave-private LDS -> A-layout fragments ----
    {
      u16a* pb = (u16a*)(pbuf + w * 16 * PROW);
      for (int c = 0; c < 4; ++c)
        for (int r = 0; r < 8; ++r)
          pb[(r + 8 * lh) * (PROW * 2) + c * 16 + l15] = f2bf(sc[c][r]);
    }
    Frag16 pa[2];
    for (int ks = 0; ks < 2; ++ks)
      for (int v = 0; v < 8; ++v) {
        const int word = l15 * PROW + ks * 16 + lh * 8 + v;
        pa[ks].u[v] = ((u32a*)pbuf)[w * 16 * PROW + word];
      }

    // ---- PV: preload both B-frags, then 2 back-to-back WMMAs ----
    for (int dt = 0; dt < 4; ++dt) {
      const int dcol = dt * 16 + l15;
      Frag16 b0, b1;
      for (int v = 0; v < 8; ++v) {
        b0.u[v] = ((u32a*)vsh)[dcol * VROW +      lh * 8 + v];
        b1.u[v] = ((u32a*)vsh)[dcol * VROW + 16 + lh * 8 + v];
      }
      o[dt] = __builtin_amdgcn_wmma_f32_16x16x32_bf16(false, pa[0].v, false, b0.v,
                                                      (short)0, o[dt], false, false);
      o[dt] = __builtin_amdgcn_wmma_f32_16x16x32_bf16(false, pa[1].v, false, b1.v,
                                                      (short)0, o[dt], false, false);
    }
  }

  // ---- normalize and store (i, b, n, d) ----
  for (int dt = 0; dt < 4; ++dt)
    for (int r = 0; r < 8; ++r) {
      const int i = qt * 64 + w * 16 + r + 8 * lh;
      const int d = dt * 16 + l15;
      out[(long long)i * rowStride + bnOff + d] = o[dt][r] / l[r];
    }
}

extern "C" void kernel_launch(void* const* d_in, const int* in_sizes, int n_in,
                              void* d_out, int out_size, void* d_ws, size_t ws_size,
                              hipStream_t stream) {
  (void)in_sizes; (void)n_in; (void)out_size; (void)d_ws; (void)ws_size;
  const float* q   = (const float*)d_in[0];
  const float* kh  = (const float*)d_in[1];
  const float* vh  = (const float*)d_in[2];
  const float* kr  = (const float*)d_in[3];
  const float* se  = (const float*)d_in[4];
  const float* sm  = (const float*)d_in[5];
  const float* rwb = (const float*)d_in[6];
  const float* rrb = (const float*)d_in[7];
  const float* rsb = (const float*)d_in[8];
  const float* am  = (const float*)d_in[9];
  dim3 grid(QLEN / 64, BSZ * NHEAD);
  relattn_flash_kernel<<<grid, dim3(128), 0, stream>>>(
      q, kh, vh, kr, se, sm, rwb, rrb, rsb, am, (float*)d_out);
}
